// Qwen3Attention_8504035246571
// MI455X (gfx1250) — matrix operand order repaired and checked by reading, not compiled, not run on hardware
//
#include <hip/hip_runtime.h>

typedef __attribute__((ext_vector_type(16))) __bf16 v16bf;
typedef __attribute__((ext_vector_type(8)))  __bf16 v8bf;
typedef __attribute__((ext_vector_type(8)))  float  v8f;

#define SEQ  2048
#define DMODEL 4096
#define NH   32
#define NKV  8
#define HD   128
#define SCALING 0.08838834764831845f  // 128^-0.5

union bfu { v16bf v; v8bf h[2]; };

__device__ __forceinline__ v8f wmma_bf16(v16bf a, v16bf b, v8f c) {
  return __builtin_amdgcn_wmma_f32_16x16x32_bf16(false, a, false, b, (short)0, c, false, false);
}

// CDNA5 async copy: global -> LDS, 16 bytes per lane, tracked by ASYNCcnt.
// NOTE: lds_dst must be a pointer derived from the __shared__ object so the
// ptrtoint capture forces the compiler to treat the asm as writing LDS.
__device__ __forceinline__ void async_b128(const void* lds_dst, const void* gaddr) {
  asm volatile("global_load_async_to_lds_b128 %0, %1, off"
               :: "v"((unsigned)(uintptr_t)lds_dst),
                  "v"((unsigned long long)(uintptr_t)gaddr)
               : "memory");
}
__device__ __forceinline__ void wait_async0() {
  asm volatile("s_wait_asynccnt 0x0" ::: "memory");
}

// A-matrix fragment (16x32 bf16) from row-major [rows][ld] buffer.
// lane<16: row=lane, K = {0..7,16..23}; lane>=16: row=lane-16, K = {8..15,24..31}.
__device__ __forceinline__ v16bf lds_a_frag(const __bf16* base, int ld, int row_base, int kbase) {
  int lane = threadIdx.x & 31;
  int r = row_base + (lane & 15);
  int half = lane >> 4;
  const __bf16* p = base + (size_t)r * ld + kbase + half * 8;
  bfu u;
  u.h[0] = *(const v8bf*)(p);
  u.h[1] = *(const v8bf*)(p + 16);
  return u.v;
}

// B-matrix fragment (32x16 bf16) from B^T layout [cols][ld] buffer (element [n][k]).
__device__ __forceinline__ v16bf lds_b_frag(const __bf16* bt, int ld, int col_base, int kbase) {
  int lane = threadIdx.x & 31;
  int c = col_base + (lane & 15);
  int half = lane >> 4;
  const __bf16* p = bt + (size_t)c * ld + kbase + half * 8;
  bfu u;
  u.h[0] = *(const v8bf*)(p);
  u.h[1] = *(const v8bf*)(p + 16);
  return u.v;
}

__device__ __forceinline__ float wave_max(float v) {
  for (int o = 16; o; o >>= 1) v = fmaxf(v, __shfl_xor(v, o, 32));
  return v;
}
__device__ __forceinline__ float wave_min(float v) {
  for (int o = 16; o; o >>= 1) v = fminf(v, __shfl_xor(v, o, 32));
  return v;
}
__device__ __forceinline__ float wave_sum(float v) {
  for (int o = 16; o; o >>= 1) v += __shfl_xor(v, o, 32);
  return v;
}

// ---------------------------------------------------------------- casts / transposes
__global__ void cast_f32_bf16(const float* __restrict__ s, __bf16* __restrict__ d, size_t n) {
  size_t i = (size_t)blockIdx.x * blockDim.x + threadIdx.x;
  size_t stride = (size_t)gridDim.x * blockDim.x;
  for (; i < n; i += stride) d[i] = (__bf16)s[i];
}

// dst[c][r] = (bf16)src[r][c]; src is f32 [R][C]. 32x32 LDS tile, block 256.
__global__ __launch_bounds__(256) void transpose_cast_f32(const float* __restrict__ src,
                                                          __bf16* __restrict__ dst,
                                                          int R, int C) {
  __shared__ __bf16 tile[32][33];
  int tx = threadIdx.x & 31, ty = threadIdx.x >> 5;
  int r0 = blockIdx.y * 32, c0 = blockIdx.x * 32;
#pragma unroll
  for (int j = 0; j < 4; ++j)
    tile[ty + 8 * j][tx] = (__bf16)src[(size_t)(r0 + ty + 8 * j) * C + c0 + tx];
  __syncthreads();
#pragma unroll
  for (int j = 0; j < 4; ++j)
    dst[(size_t)(c0 + ty + 8 * j) * R + r0 + tx] = tile[tx][ty + 8 * j];
}

// batched bf16 transpose: dst[z][c][r] = src[z][r][c]
__global__ __launch_bounds__(256) void transpose_bf16(const __bf16* __restrict__ src,
                                                      __bf16* __restrict__ dst,
                                                      int R, int C) {
  __shared__ __bf16 tile[32][33];
  const __bf16* s = src + (size_t)blockIdx.z * R * C;
  __bf16* d = dst + (size_t)blockIdx.z * R * C;
  int tx = threadIdx.x & 31, ty = threadIdx.x >> 5;
  int r0 = blockIdx.y * 32, c0 = blockIdx.x * 32;
#pragma unroll
  for (int j = 0; j < 4; ++j)
    tile[ty + 8 * j][tx] = s[(size_t)(r0 + ty + 8 * j) * C + c0 + tx];
  __syncthreads();
#pragma unroll
  for (int j = 0; j < 4; ++j)
    d[(size_t)(c0 + ty + 8 * j) * R + r0 + tx] = tile[tx][ty + 8 * j];
}

// ---------------------------------------------------------------- GEMM
// C[M,N] = A[M,K] @ B[K,N] with B supplied pre-transposed as Bt[N][K] (bf16).
// Block tile 128x128, BK=32, 8 waves of 32x64. Async double-buffered LDS staging.
#define GBM 128
#define GBN 128
#define GBK 32
__global__ __launch_bounds__(256) void gemm_bf16_wmma(const __bf16* __restrict__ A,
                                                      const __bf16* __restrict__ Bt,
                                                      float* __restrict__ C,
                                                      int M, int N, int K) {
  // buf0: As @0 (8KB), Bts @8192 (8KB); buf1: @16384 / @24576
  __shared__ __attribute__((aligned(32))) char gsmem[32768];
  int tid = threadIdx.x, lane = tid & 31, wave = tid >> 5;
  int m0 = blockIdx.y * GBM, n0 = blockIdx.x * GBN;
  int mw = wave & 3, nw = wave >> 2;  // wave tile origin (mw*32, nw*64)

  // staging: thread -> one row (of 128), 16 contiguous bf16 (2x b128)
  int srow = tid >> 1, sseg = (tid & 1) * 16;
  const __bf16* aptr = A + (size_t)(m0 + srow) * K + sseg;
  const __bf16* bptr = Bt + (size_t)(n0 + srow) * K + sseg;
  char* adst = gsmem + srow * 64 + sseg * 2;
  char* bdst = gsmem + 8192 + srow * 64 + sseg * 2;

  v8f acc[2][4] = {};

  // preload K-tile 0 into buffer 0
  async_b128(adst, aptr);
  async_b128(adst + 16, aptr + 8);
  async_b128(bdst, bptr);
  async_b128(bdst + 16, bptr + 8);

  int T = K / GBK;
  for (int it = 0; it < T; ++it) {
    unsigned base = (it & 1) ? 16384u : 0u;
    wait_async0();
    __syncthreads();  // tile 'it' resident; previous tile fully consumed by all waves
    if (it + 1 < T) {
      unsigned nb = (it & 1) ? 0u : 16384u;
      const __bf16* an = aptr + (size_t)(it + 1) * GBK;
      const __bf16* bn = bptr + (size_t)(it + 1) * GBK;
      async_b128(adst + nb, an);
      async_b128(adst + nb + 16, an + 8);
      async_b128(bdst + nb, bn);
      async_b128(bdst + nb + 16, bn + 8);
    }
    const __bf16* As = (const __bf16*)(gsmem + base);
    const __bf16* Bs = (const __bf16*)(gsmem + base + 8192);
#pragma unroll
    for (int i = 0; i < 2; ++i) {
      v16bf a = lds_a_frag(As, GBK, mw * 32 + i * 16, 0);
#pragma unroll
      for (int j = 0; j < 4; ++j) {
        v16bf b = lds_b_frag(Bs, GBK, nw * 64 + j * 16, 0);
        acc[i][j] = wmma_bf16(a, b, acc[i][j]);
      }
    }
  }
  int colh = lane & 15, rh = (lane >> 4) * 8;
#pragma unroll
  for (int i = 0; i < 2; ++i)
#pragma unroll
    for (int j = 0; j < 4; ++j) {
      int row = m0 + mw * 32 + i * 16 + rh;
      int col = n0 + nw * 64 + j * 16 + colh;
#pragma unroll
      for (int e = 0; e < 8; ++e) C[(size_t)(row + e) * N + col] = acc[i][j][e];
    }
}

// ---------------------------------------------------------------- RMSNorm + RoPE + transpose
__global__ __launch_bounds__(128) void norm_rope_kernel(const float* __restrict__ qf,
                                                        const float* __restrict__ kf,
                                                        const float* __restrict__ vf,
                                                        const float* __restrict__ qw,
                                                        const float* __restrict__ kw,
                                                        const float* __restrict__ cosb,
                                                        const float* __restrict__ sinb,
                                                        __bf16* __restrict__ qb,
                                                        __bf16* __restrict__ kb,
                                                        __bf16* __restrict__ vb) {
  int s = blockIdx.x, j = blockIdx.y, t = threadIdx.x;
  __shared__ float y[HD];
  __shared__ float red[HD];

  if (j < NH + NKV) {
    bool isq = (j < NH);
    int head = isq ? j : (j - NH);
    float x = isq ? qf[(size_t)s * DMODEL + head * HD + t]
                  : kf[(size_t)s * (NKV * HD) + head * HD + t];
    red[t] = x * x;
    __syncthreads();
    for (int o = 64; o; o >>= 1) {
      if (t < o) red[t] += red[t + o];
      __syncthreads();
    }
    float var = red[0] * (1.0f / 128.0f);
    float w = isq ? qw[t] : kw[t];
    float yn = x * rsqrtf(var + 1e-6f) * w;
    y[t] = yn;
    __syncthreads();
    float rot = (t < 64) ? -y[t + 64] : y[t - 64];
    float c = cosb[(size_t)s * HD + t], sn = sinb[(size_t)s * HD + t];
    float out = yn * c + rot * sn;
    __bf16* dst = isq ? qb : kb;
    dst[((size_t)head * SEQ + s) * HD + t] = (__bf16)out;
  } else {
    int kvh = j - NH - NKV;
    vb[((size_t)kvh * SEQ + s) * HD + t] =
        (__bf16)vf[(size_t)s * (NKV * HD) + kvh * HD + t];
  }
}

// ---------------------------------------------------------------- fused attention
// grid (S/32, H), block 256 (8 waves). V supplied pre-transposed: vt[kv][hd][SEQ].
__global__ __launch_bounds__(256) void attn_kernel(const __bf16* __restrict__ qb,
                                                   const __bf16* __restrict__ kb,
                                                   const __bf16* __restrict__ vt,
                                                   __bf16* __restrict__ aob) {
  extern __shared__ char smem[];
  __bf16* Qs   = (__bf16*)smem;               // [32][128]  8 KB  @ 0
  __bf16* KV   = (__bf16*)(smem + 8192);      // [128][128] 32 KB @ 8192 (K tile, then V^T tile)
  float*  Sc   = (float*)(smem + 40960);      // [32][128] 16 KB
  __bf16* Ps   = (__bf16*)(smem + 57344);     // [32][128]  8 KB
  float*  mrow = (float*)(smem + 65536);      // [32]
  float*  lrow = (float*)(smem + 65536 + 128);

  int tid = threadIdx.x, lane = tid & 31, wave = tid >> 5;
  int h = blockIdx.y, kvh = h >> 2;
  int q0 = blockIdx.x * 32;
  int mw = wave & 1, nw = wave >> 1, m_base = mw * 16;

  { // stage Q block 32x128 via async (row stride 256B in LDS)
    const __bf16* src = qb + ((size_t)h * SEQ + q0) * HD;
    int row = tid >> 3, seg = (tid & 7) * 16;
    char* qdst = smem + row * 256 + seg * 2;
    async_b128(qdst, src + (size_t)row * HD + seg);
    async_b128(qdst + 16, src + (size_t)row * HD + seg + 8);
  }
  if (tid < 32) { mrow[tid] = -1e30f; lrow[tid] = 0.0f; }

  int ktiles = (q0 + 31) / 128 + 1;
  int krow = tid >> 1, kseg = (tid & 1) * 64;  // staging coords for 128x128 tiles
  char* kvdst = smem + 8192 + krow * 256 + kseg * 2;

  // ---------------- Pass A: row max + sum(exp) ----------------
  for (int kt = 0; kt < ktiles; ++kt) {
    { // K tile [key][hd], contiguous: 128B per thread = 8x b128
      const __bf16* ks = kb + ((size_t)kvh * SEQ + kt * 128 + krow) * HD + kseg;
#pragma unroll
      for (int c = 0; c < 8; ++c) async_b128(kvdst + c * 16, ks + c * 8);
    }
    wait_async0();
    __syncthreads();
#pragma unroll
    for (int j = 0; j < 2; ++j) {
      v8f acc = {};
#pragma unroll
      for (int kc = 0; kc < 4; ++kc) {
        v16bf a = lds_a_frag(Qs, HD, m_base, kc * 32);
        v16bf b = lds_b_frag(KV, HD, nw * 32 + j * 16, kc * 32);
        acc = wmma_bf16(a, b, acc);
      }
      int col = nw * 32 + j * 16 + (lane & 15);
      int rh = (lane >> 4) * 8;
#pragma unroll
      for (int e = 0; e < 8; ++e) Sc[(m_base + rh + e) * 128 + col] = acc[e];
    }
    __syncthreads();
#pragma unroll
    for (int rr = 0; rr < 4; ++rr) {
      int r = wave * 4 + rr, qrow = q0 + r;
      float vals[4], tmax = -1e30f;
#pragma unroll
      for (int c4 = 0; c4 < 4; ++c4) {
        int col = c4 * 32 + lane;
        float sv = Sc[r * 128 + col] * SCALING;
        if (kt * 128 + col > qrow) sv = -1e30f;
        vals[c4] = sv;
        tmax = fmaxf(tmax, sv);
      }
      tmax = wave_max(tmax);
      float mold = mrow[r], lold = lrow[r];
      float mnew = fmaxf(mold, tmax);
      float psum = 0.0f;
#pragma unroll
      for (int c4 = 0; c4 < 4; ++c4) psum += __expf(vals[c4] - mnew);
      psum = wave_sum(psum);
      if (lane == 0) { mrow[r] = mnew; lrow[r] = lold * __expf(mold - mnew) + psum; }
    }
    __syncthreads();
  }

  // ---------------- Pass B: probs + group fake-quant + P@V ----------------
  v8f o0 = {}, o1 = {};
  for (int kt = 0; kt < ktiles; ++kt) {
    {
      const __bf16* ks = kb + ((size_t)kvh * SEQ + kt * 128 + krow) * HD + kseg;
#pragma unroll
      for (int c = 0; c < 8; ++c) async_b128(kvdst + c * 16, ks + c * 8);
    }
    wait_async0();
    __syncthreads();
#pragma unroll
    for (int j = 0; j < 2; ++j) {
      v8f acc = {};
#pragma unroll
      for (int kc = 0; kc < 4; ++kc) {
        v16bf a = lds_a_frag(Qs, HD, m_base, kc * 32);
        v16bf b = lds_b_frag(KV, HD, nw * 32 + j * 16, kc * 32);
        acc = wmma_bf16(a, b, acc);
      }
      int col = nw * 32 + j * 16 + (lane & 15);
      int rh = (lane >> 4) * 8;
#pragma unroll
      for (int e = 0; e < 8; ++e) Sc[(m_base + rh + e) * 128 + col] = acc[e];
    }
    __syncthreads();
    // probs + 8-bit group fake-quant (group == this 128-key tile, GS=128)
#pragma unroll
    for (int rr = 0; rr < 4; ++rr) {
      int r = wave * 4 + rr, qrow = q0 + r;
      float m = mrow[r], linv = 1.0f / lrow[r];
      float p[4], pmax = -1e30f, pmin = 1e30f;
#pragma unroll
      for (int c4 = 0; c4 < 4; ++c4) {
        int col = c4 * 32 + lane;
        float sv = Sc[r * 128 + col] * SCALING;
        if (kt * 128 + col > qrow) sv = -1e30f;
        float pp = __expf(sv - m) * linv;
        p[c4] = pp;
        pmax = fmaxf(pmax, pp);
        pmin = fminf(pmin, pp);
      }
      pmax = wave_max(pmax);
      pmin = wave_min(pmin);
      float qscale = fmaxf((pmax - pmin) * (1.0f / 255.0f), 1e-12f);
      float zero = rintf(-pmin / qscale);
#pragma unroll
      for (int c4 = 0; c4 < 4; ++c4) {
        int col = c4 * 32 + lane;
        float qv = rintf(p[c4] / qscale) + zero;
        qv = fminf(fmaxf(qv, 0.0f), 255.0f);
        Ps[r * 128 + col] = (__bf16)(qscale * (qv - zero));
      }
    }
    __syncthreads();
    { // stage V^T tile Vt[hd][key] from pre-transposed vt[kv][hd][SEQ]: contiguous async
      const __bf16* vs = vt + ((size_t)kvh * HD + krow) * SEQ + kt * 128 + kseg;
#pragma unroll
      for (int c = 0; c < 8; ++c) async_b128(kvdst + c * 16, vs + c * 8);
    }
    wait_async0();
    __syncthreads();
#pragma unroll
    for (int kc = 0; kc < 4; ++kc) {
      v16bf a = lds_a_frag(Ps, 128, m_base, kc * 32);
      v16bf b0 = lds_b_frag(KV, 128, nw * 32, kc * 32);
      o0 = wmma_bf16(a, b0, o0);
      v16bf b1 = lds_b_frag(KV, 128, nw * 32 + 16, kc * 32);
      o1 = wmma_bf16(a, b1, o1);
    }
    __syncthreads();  // KV free before next tile's async stores
  }

  int colh = lane & 15, rh = (lane >> 4) * 8;
  int rbase = q0 + m_base + rh;
#pragma unroll
  for (int e = 0; e < 8; ++e) {
    aob[(size_t)(rbase + e) * DMODEL + h * HD + nw * 32 + colh]      = (__bf16)o0[e];
    aob[(size_t)(rbase + e) * DMODEL + h * HD + nw * 32 + 16 + colh] = (__bf16)o1[e];
  }
}

// ----------------------------------------------------------------
extern "C" void kernel_launch(void* const* d_in, const int* in_sizes, int n_in,
                              void* d_out, int out_size, void* d_ws, size_t ws_size,
                              hipStream_t stream) {
  const float* hs   = (const float*)d_in[0];
  const float* wq   = (const float*)d_in[1];
  const float* wk   = (const float*)d_in[2];
  const float* wv   = (const float*)d_in[3];
  const float* wo   = (const float*)d_in[4];
  const float* qnw  = (const float*)d_in[5];
  const float* knw  = (const float*)d_in[6];
  const float* cosb = (const float*)d_in[7];
  const float* sinb = (const float*)d_in[8];
  float* out = (float*)d_out;

  char* w = (char*)d_ws;
  size_t off = 0;
  auto alloc = [&](size_t bytes) {
    char* p = w + off;
    off += (bytes + 255) & ~(size_t)255;
    return p;
  };
  __bf16* hsb  = (__bf16*)alloc((size_t)SEQ * DMODEL * 2);
  __bf16* wqT  = (__bf16*)alloc((size_t)DMODEL * DMODEL * 2);   // [N][K]
  __bf16* wkT  = (__bf16*)alloc((size_t)DMODEL * NKV * HD * 2);
  __bf16* wvT  = (__bf16*)alloc((size_t)DMODEL * NKV * HD * 2);
  __bf16* woT  = (__bf16*)alloc((size_t)DMODEL * DMODEL * 2);
  float*  qf   = (float*)alloc((size_t)SEQ * DMODEL * 4);
  float*  kf   = (float*)alloc((size_t)SEQ * NKV * HD * 4);
  float*  vf   = (float*)alloc((size_t)SEQ * NKV * HD * 4);
  __bf16* qbuf = (__bf16*)alloc((size_t)NH * SEQ * HD * 2);
  __bf16* kbuf = (__bf16*)alloc((size_t)NKV * SEQ * HD * 2);
  __bf16* vbuf = (__bf16*)alloc((size_t)NKV * SEQ * HD * 2);
  __bf16* aob  = (__bf16*)qf;  // qf dead after norm_rope
  __bf16* vbT  = (__bf16*)kf;  // kf dead after norm_rope; 4MB <= 8MB

  cast_f32_bf16<<<4096, 256, 0, stream>>>(hs, hsb, (size_t)SEQ * DMODEL);
  // weights: fused cast + transpose -> Bt[N][K]
  transpose_cast_f32<<<dim3(DMODEL / 32, DMODEL / 32), 256, 0, stream>>>(wq, wqT, DMODEL, DMODEL);
  transpose_cast_f32<<<dim3(NKV * HD / 32, DMODEL / 32), 256, 0, stream>>>(wk, wkT, DMODEL, NKV * HD);
  transpose_cast_f32<<<dim3(NKV * HD / 32, DMODEL / 32), 256, 0, stream>>>(wv, wvT, DMODEL, NKV * HD);
  transpose_cast_f32<<<dim3(DMODEL / 32, DMODEL / 32), 256, 0, stream>>>(wo, woT, DMODEL, DMODEL);

  gemm_bf16_wmma<<<dim3(DMODEL / GBN, SEQ / GBM), 256, 0, stream>>>(hsb, wqT, qf, SEQ, DMODEL, DMODEL);
  gemm_bf16_wmma<<<dim3(NKV * HD / GBN, SEQ / GBM), 256, 0, stream>>>(hsb, wkT, kf, SEQ, NKV * HD, DMODEL);
  gemm_bf16_wmma<<<dim3(NKV * HD / GBN, SEQ / GBM), 256, 0, stream>>>(hsb, wvT, vf, SEQ, NKV * HD, DMODEL);

  norm_rope_kernel<<<dim3(SEQ, NH + NKV + NKV), 128, 0, stream>>>(
      qf, kf, vf, qnw, knw, cosb, sinb, qbuf, kbuf, vbuf);
  // V: [kv][s][hd] -> [kv][hd][s] for contiguous V^T staging in attention
  transpose_bf16<<<dim3(HD / 32, SEQ / 32, NKV), 256, 0, stream>>>(vbuf, vbT, SEQ, HD);

  int smem_bytes = 65536 + 256;
  hipFuncSetAttribute((const void*)attn_kernel,
                      hipFuncAttributeMaxDynamicSharedMemorySize, smem_bytes);
  attn_kernel<<<dim3(SEQ / 32, NH), 256, smem_bytes, stream>>>(qbuf, kbuf, vbT, aob);

  gemm_bf16_wmma<<<dim3(DMODEL / GBN, SEQ / GBM), 256, 0, stream>>>(aob, woT, out, SEQ, DMODEL, DMODEL);
}
